// BiPixelMambaLayer_5617817223710
// MI455X (gfx1250) — matrix operand order repaired and checked by reading, not compiled, not run on hardware
//
#include <hip/hip_runtime.h>

#define DIMC   96
#define DSTATE 16
#define DCONV  4
#define DINNER 192
#define DTRANK 6
#define BATCH  8
#define LSEQ   2304                 // 48*48
#define MTOK   (BATCH*LSEQ)         // 18432
#define NPROJ  (2*DINNER)           // 384
#define NDBL   48                   // 38 padded to 48 for 16x16 tiles
#define NSEG   16
#define SEGLEN (LSEQ/NSEG)          // 144
#define NDT    (DINNER/16)          // 12 d-tiles

typedef __attribute__((ext_vector_type(16))) __bf16 v16bf;
typedef __attribute__((ext_vector_type(8)))  __bf16 v8bf;
typedef __attribute__((ext_vector_type(8)))  float  v8f;

__device__ __forceinline__ float sigmoidf_(float x) { return 1.f / (1.f + __expf(-x)); }
__device__ __forceinline__ float softplusf_(float x) {
  return (x > 20.f) ? x : log1pf(__expf(x));
}

// ---------------- LayerNorm over channels: x (B,C,L) -> xn (M,C) f32 + bf16 ----
__global__ __launch_bounds__(256) void k_layernorm(const float* __restrict__ x,
                                                   const float* __restrict__ g,
                                                   const float* __restrict__ bb,
                                                   float* __restrict__ xn,
                                                   __bf16* __restrict__ xnbf) {
  int m = blockIdx.x * 256 + threadIdx.x;
  if (m >= MTOK) return;
  int b = m / LSEQ, l = m - b * LSEQ;
  const float* xb = x + (size_t)b * DIMC * LSEQ + l;   // coalesced over l across threads
  float s = 0.f, s2 = 0.f;
  for (int c = 0; c < DIMC; ++c) { float v = xb[(size_t)c * LSEQ]; s += v; s2 += v * v; }
  float mu = s * (1.f / DIMC);
  float var = s2 * (1.f / DIMC) - mu * mu;
  float rs = rsqrtf(var + 1e-5f);
  float* od = xn + (size_t)m * DIMC;
  __bf16* ob = xnbf + (size_t)m * DIMC;
  for (int c = 0; c < DIMC; ++c) {
    float v = (xb[(size_t)c * LSEQ] - mu) * rs * g[c] + bb[c];
    od[c] = v;
    ob[c] = (__bf16)v;
  }
}

// ---------------- weight fp32 -> bf16 (with optional zero row padding) --------
__global__ __launch_bounds__(256) void k_w2bf(const float* __restrict__ src,
                                              __bf16* __restrict__ dst,
                                              int rows_src, int rows_dst, int cols) {
  int i = blockIdx.x * 256 + threadIdx.x;
  int n = rows_dst * cols;
  if (i >= n) return;
  int r = i / cols, c = i - r * cols;
  dst[i] = (r < rows_src) ? (__bf16)src[r * cols + c] : (__bf16)0.f;
}

// ---------------- generic bf16 WMMA GEMM: C[M,N](ldc,f32) = A[M,K] * B[N,K]^T -
__global__ __launch_bounds__(128) void k_gemm_bf16(const __bf16* __restrict__ A,
                                                   const __bf16* __restrict__ Bw,
                                                   float* __restrict__ C,
                                                   int N, int K, int ldc) {
  int wave = threadIdx.x >> 5;
  int lane = threadIdx.x & 31;
  int r = lane & 15, hi = lane >> 4;
  int m0 = blockIdx.x * 16;
  int n0 = (blockIdx.y * 4 + wave) * 16;
  if (n0 >= N) return;                        // wave-uniform; EXEC stays all-1s for WMMA
  v8f acc = {};
  const __bf16* arow = A + (size_t)(m0 + r) * K;
  const __bf16* brow = Bw + (size_t)(n0 + r) * K;
  for (int kk = 0; kk < K; kk += 32) {
    // A 16x32 bf16: lanes0-15 K kk..+7 / kk+16..+23; lanes16-31 K kk+8..+15 / kk+24..+31
    v8bf alo = *(const v8bf*)(arow + kk + hi * 8);
    v8bf ahi = *(const v8bf*)(arow + kk + 16 + hi * 8);
    v16bf a = __builtin_shufflevector(alo, ahi, 0,1,2,3,4,5,6,7,8,9,10,11,12,13,14,15);
    // B 32x16 bf16: same K order as A (lanes0-15 K kk..+7 / kk+16..+23; lanes16-31 K kk+8..+15 / kk+24..+31)
    v8bf blo = *(const v8bf*)(brow + kk + hi * 8);
    v8bf bhi = *(const v8bf*)(brow + kk + 16 + hi * 8);
    v16bf bm = __builtin_shufflevector(blo, bhi, 0,1,2,3,4,5,6,7,8,9,10,11,12,13,14,15);
    acc = __builtin_amdgcn_wmma_f32_16x16x32_bf16(false, a, false, bm, (short)0, acc,
                                                  false, false);
  }
  float* crow = C + (size_t)(m0 + hi * 8) * ldc + n0 + r;
#pragma unroll
  for (int v = 0; v < 8; ++v) crow[(size_t)v * ldc] = acc[v];
}

// ---------------- depthwise causal conv (dir-mirrored taps) + bias + SiLU -----
__global__ __launch_bounds__(256) void k_conv_silu(const float* __restrict__ xz,
                                                   const float* __restrict__ w,
                                                   const float* __restrict__ bias,
                                                   int dir,
                                                   float* __restrict__ xcf,
                                                   __bf16* __restrict__ xcb) {
  int i = blockIdx.x * 256 + threadIdx.x;
  if (i >= MTOK * DINNER) return;
  int d = i % DINNER;
  int m = i / DINNER;
  int l = m % LSEQ;
  float acc = bias[d];
#pragma unroll
  for (int k = 0; k < DCONV; ++k) {
    int off = dir ? (DCONV - 1 - k) : (k - (DCONV - 1));
    int ln = l + off;
    if (ln >= 0 && ln < LSEQ) acc += w[d * DCONV + k] * xz[(size_t)(m + off) * NPROJ + d];
  }
  acc = acc * sigmoidf_(acc);                 // SiLU
  xcf[(size_t)m * DINNER + d] = acc;
  xcb[(size_t)m * DINNER + d] = (__bf16)acc;
}

// =====================  segmented linear scan (3 passes)  =====================
// h_t = exp(delta_t*A)*h_{t-1} + delta_t*u_t*B_t  is linear in h, so per segment:
//   h_seg_end = a_seg*h0 + b_seg   with a_seg = exp(A * sum(delta))
// Pass1: per-segment (a_seg, h_end from h0=0), parallel over 1536 blocks.
// Pass2: serial stitch over NSEG=16 segments (96 blocks).
// Pass3: rerun segments from correct h0, emit gated y.

// Pass 1
__global__ __launch_bounds__(256) void k_scan_seg(const float* __restrict__ u,
                                                  const float* __restrict__ dbl,
                                                  const float* __restrict__ dtw,
                                                  const float* __restrict__ dtb,
                                                  const float* __restrict__ Alog,
                                                  int dir,
                                                  float* __restrict__ hend,
                                                  float* __restrict__ aseg) {
  int s = blockIdx.x % NSEG;
  int bd = blockIdx.x / NSEG;
  int b = bd / NDT, dt = bd % NDT;
  int dl = threadIdx.x >> 4;
  int n = threadIdx.x & 15;
  int d = dt * 16 + dl;
  float A = -__expf(Alog[d * DSTATE + n]);
  float w0 = dtw[d * DTRANK + 0], w1 = dtw[d * DTRANK + 1], w2 = dtw[d * DTRANK + 2];
  float w3 = dtw[d * DTRANK + 3], w4 = dtw[d * DTRANK + 4], w5 = dtw[d * DTRANK + 5];
  float bd_ = dtb[d];
  float h = 0.f, dsum = 0.f;
  for (int t = s * SEGLEN; t < (s + 1) * SEGLEN; ++t) {
    int l = dir ? (LSEQ - 1 - t) : t;
    size_t m = (size_t)b * LSEQ + l;
    const float* db = dbl + m * NDBL;
    float dp = db[0]*w0 + db[1]*w1 + db[2]*w2 + db[3]*w3 + db[4]*w4 + db[5]*w5 + bd_;
    float delta = softplusf_(dp);
    float ut = u[m * DINNER + d];
    float Bt = db[DTRANK + n];
    h = __expf(delta * A) * h + (delta * ut) * Bt;
    dsum += delta;
  }
  size_t idx = (((size_t)b * NSEG + s) * DINNER + d) * DSTATE + n;
  hend[idx] = h;
  aseg[idx] = __expf(dsum * A);
}

// Pass 2: sequential over the 16 segments (cheap), emits per-segment initial h
__global__ __launch_bounds__(256) void k_scan_stitch(const float* __restrict__ hend,
                                                     const float* __restrict__ aseg,
                                                     float* __restrict__ hinit) {
  int b = blockIdx.x / NDT, dt = blockIdx.x % NDT;
  int dl = threadIdx.x >> 4;
  int n = threadIdx.x & 15;
  int d = dt * 16 + dl;
  float h = 0.f;
  for (int s = 0; s < NSEG; ++s) {
    size_t idx = (((size_t)b * NSEG + s) * DINNER + d) * DSTATE + n;
    hinit[idx] = h;
    h = aseg[idx] * h + hend[idx];
  }
}

// Pass 3: rerun from correct h0, 16-lane shfl reduction over state, gate & emit
__global__ __launch_bounds__(256) void k_scan_emit(const float* __restrict__ u,
                                                   const float* __restrict__ dbl,
                                                   const float* __restrict__ xz,
                                                   const float* __restrict__ dtw,
                                                   const float* __restrict__ dtb,
                                                   const float* __restrict__ Alog,
                                                   const float* __restrict__ Dp,
                                                   const float* __restrict__ hinit,
                                                   int dir,
                                                   float* __restrict__ yacc,
                                                   __bf16* __restrict__ ybf) {
  int s = blockIdx.x % NSEG;
  int bd = blockIdx.x / NSEG;
  int b = bd / NDT, dt = bd % NDT;
  int dl = threadIdx.x >> 4;
  int n = threadIdx.x & 15;
  int d = dt * 16 + dl;
  float A = -__expf(Alog[d * DSTATE + n]);
  float w0 = dtw[d * DTRANK + 0], w1 = dtw[d * DTRANK + 1], w2 = dtw[d * DTRANK + 2];
  float w3 = dtw[d * DTRANK + 3], w4 = dtw[d * DTRANK + 4], w5 = dtw[d * DTRANK + 5];
  float bd_ = dtb[d], Dd = Dp[d];
  float h = hinit[(((size_t)b * NSEG + s) * DINNER + d) * DSTATE + n];
  for (int t = s * SEGLEN; t < (s + 1) * SEGLEN; ++t) {
    int l = dir ? (LSEQ - 1 - t) : t;
    size_t m = (size_t)b * LSEQ + l;
    const float* db = dbl + m * NDBL;
    float dp = db[0]*w0 + db[1]*w1 + db[2]*w2 + db[3]*w3 + db[4]*w4 + db[5]*w5 + bd_;
    float delta = softplusf_(dp);
    float ut = u[m * DINNER + d];
    float Bt = db[DTRANK + n];
    float Ct = db[DTRANK + DSTATE + n];
    h = __expf(delta * A) * h + (delta * ut) * Bt;
    float y = h * Ct;
    y += __shfl_xor(y, 8, 32);
    y += __shfl_xor(y, 4, 32);
    y += __shfl_xor(y, 2, 32);
    y += __shfl_xor(y, 1, 32);
    if (n == 0) {
      float z = xz[m * NPROJ + DINNER + d];
      float yt = (y + ut * Dd) * (z * sigmoidf_(z));
      if (dir == 0) yacc[m * DINNER + d] = yt;
      else          ybf[m * DINNER + d] = (__bf16)(yacc[m * DINNER + d] + yt);
    }
  }
}

// ---------------- output GEMM (y @ out_w^T) + residual, store as (B,C,L) ------
__global__ __launch_bounds__(128) void k_gemm_out(const __bf16* __restrict__ Y,
                                                  const __bf16* __restrict__ Wo,
                                                  const float* __restrict__ x,
                                                  float* __restrict__ out) {
  int wave = threadIdx.x >> 5;
  int lane = threadIdx.x & 31;
  int r = lane & 15, hi = lane >> 4;
  int m0 = blockIdx.x * 16;
  int n0 = (blockIdx.y * 4 + wave) * 16;
  if (n0 >= DIMC) return;
  v8f acc = {};
  const __bf16* arow = Y + (size_t)(m0 + r) * DINNER;
  const __bf16* brow = Wo + (size_t)(n0 + r) * DINNER;
  for (int kk = 0; kk < DINNER; kk += 32) {
    v8bf alo = *(const v8bf*)(arow + kk + hi * 8);
    v8bf ahi = *(const v8bf*)(arow + kk + 16 + hi * 8);
    v16bf a = __builtin_shufflevector(alo, ahi, 0,1,2,3,4,5,6,7,8,9,10,11,12,13,14,15);
    v8bf blo = *(const v8bf*)(brow + kk + hi * 8);
    v8bf bhi = *(const v8bf*)(brow + kk + 16 + hi * 8);
    v16bf bm = __builtin_shufflevector(blo, bhi, 0,1,2,3,4,5,6,7,8,9,10,11,12,13,14,15);
    acc = __builtin_amdgcn_wmma_f32_16x16x32_bf16(false, a, false, bm, (short)0, acc,
                                                  false, false);
  }
  int c = n0 + r;
  int mbase = m0 + hi * 8;                    // LSEQ % 16 == 0 -> rows stay in one batch
  int b = mbase / LSEQ;
  int lb = mbase - b * LSEQ;
  const float* xr = x + ((size_t)b * DIMC + c) * LSEQ + lb;
  float* orow = out + ((size_t)b * DIMC + c) * LSEQ + lb;
#pragma unroll
  for (int v = 0; v < 8; ++v) orow[v] = acc[v] + xr[v];   // consecutive l: coalesced
}

extern "C" void kernel_launch(void* const* d_in, const int* in_sizes, int n_in,
                              void* d_out, int out_size, void* d_ws, size_t ws_size,
                              hipStream_t stream) {
  (void)in_sizes; (void)n_in; (void)out_size; (void)ws_size;
  const float* x    = (const float*)d_in[0];
  const float* ln_g = (const float*)d_in[1];
  const float* ln_b = (const float*)d_in[2];
  const float* outw = (const float*)d_in[3];
  const float* in_w[2]   = {(const float*)d_in[4],  (const float*)d_in[12]};
  const float* conv_w[2] = {(const float*)d_in[5],  (const float*)d_in[13]};
  const float* conv_b[2] = {(const float*)d_in[6],  (const float*)d_in[14]};
  const float* xproj[2]  = {(const float*)d_in[7],  (const float*)d_in[15]};
  const float* dtw[2]    = {(const float*)d_in[8],  (const float*)d_in[16]};
  const float* dtb[2]    = {(const float*)d_in[9],  (const float*)d_in[17]};
  const float* Alog[2]   = {(const float*)d_in[10], (const float*)d_in[18]};
  const float* Dp[2]     = {(const float*)d_in[11], (const float*)d_in[19]};
  float* out = (float*)d_out;

  char* ws = (char*)d_ws;
  size_t off = 0;
  auto take = [&](size_t bytes) -> char* {
    char* p = ws + off;
    off = (off + bytes + 255) & ~(size_t)255;
    return p;
  };
  float*  xn    = (float*)take((size_t)MTOK * DIMC * 4);
  __bf16* xnbf  = (__bf16*)take((size_t)MTOK * DIMC * 2);
  __bf16* inwbf = (__bf16*)take((size_t)NPROJ * DIMC * 2);
  __bf16* xpbf  = (__bf16*)take((size_t)NDBL * DINNER * 2);
  __bf16* owbf  = (__bf16*)take((size_t)DIMC * DINNER * 2);
  float*  xz    = (float*)take((size_t)MTOK * NPROJ * 4);
  float*  xcf   = (float*)take((size_t)MTOK * DINNER * 4);
  __bf16* xcb   = (__bf16*)take((size_t)MTOK * DINNER * 2);
  float*  dbl   = (float*)take((size_t)MTOK * NDBL * 4);
  float*  yacc  = (float*)take((size_t)MTOK * DINNER * 4);
  __bf16* ybf   = (__bf16*)take((size_t)MTOK * DINNER * 2);
  size_t  segN  = (size_t)BATCH * NSEG * DINNER * DSTATE;
  float*  hend  = (float*)take(segN * 4);
  float*  aseg  = (float*)take(segN * 4);
  float*  hinit = (float*)take(segN * 4);

  k_layernorm<<<MTOK / 256, 256, 0, stream>>>(x, ln_g, ln_b, xn, xnbf);
  {
    int n = DIMC * DINNER;
    k_w2bf<<<(n + 255) / 256, 256, 0, stream>>>(outw, owbf, DIMC, DIMC, DINNER);
  }
  for (int dir = 0; dir < 2; ++dir) {
    {
      int n = NPROJ * DIMC;
      k_w2bf<<<(n + 255) / 256, 256, 0, stream>>>(in_w[dir], inwbf, NPROJ, NPROJ, DIMC);
    }
    {
      int n = NDBL * DINNER;
      k_w2bf<<<(n + 255) / 256, 256, 0, stream>>>(xproj[dir], xpbf, DTRANK + 2 * DSTATE,
                                                  NDBL, DINNER);
    }
    // xz = xn @ in_w^T (per-token, direction-order independent)
    k_gemm_bf16<<<dim3(MTOK / 16, NPROJ / 64), 128, 0, stream>>>(xnbf, inwbf, xz,
                                                                 NPROJ, DIMC, NPROJ);
    {
      int n = MTOK * DINNER;
      k_conv_silu<<<(n + 255) / 256, 256, 0, stream>>>(xz, conv_w[dir], conv_b[dir], dir,
                                                       xcf, xcb);
    }
    // dbl = xc @ xproj_w^T (padded N=48)
    k_gemm_bf16<<<dim3(MTOK / 16, (NDBL + 63) / 64), 128, 0, stream>>>(xcb, xpbf, dbl,
                                                                       NDBL, DINNER, NDBL);
    // segmented scan: parallel segment pass, serial stitch, parallel emit
    k_scan_seg<<<BATCH * NDT * NSEG, 256, 0, stream>>>(xcf, dbl, dtw[dir], dtb[dir],
                                                       Alog[dir], dir, hend, aseg);
    k_scan_stitch<<<BATCH * NDT, 256, 0, stream>>>(hend, aseg, hinit);
    k_scan_emit<<<BATCH * NDT * NSEG, 256, 0, stream>>>(xcf, dbl, xz, dtw[dir], dtb[dir],
                                                        Alog[dir], Dp[dir], hinit, dir,
                                                        yacc, ybf);
  }
  k_gemm_out<<<dim3(MTOK / 16, 2), 128, 0, stream>>>(ybf, owbf, x, out);
}
